// SelfAttention_17428977287762
// MI455X (gfx1250) — matrix operand order repaired and checked by reading, not compiled, not run on hardware
//
#include <hip/hip_runtime.h>
#include <cstddef>
#include <cstdint>

// Problem constants (match reference)
static constexpr int BN  = 2;
static constexpr int SS  = 2048;
static constexpr int DD  = 2048;
static constexpr int HH  = 16;
static constexpr int HDD = 128;   // head dim
static constexpr int RR  = 32;    // LoRA rank
static constexpr int N3  = 3 * DD;        // 6144
static constexpr int MM  = BN * SS;       // 4096 rows

typedef __attribute__((ext_vector_type(16))) _Float16 v16h;
typedef __attribute__((ext_vector_type(8)))  float    v8f;

union Frag {
    uint4 u[2];   // 32 bytes = 16 halfs
    v16h  v;
};

static __device__ __forceinline__ v8f wmma16(const Frag& a, const Frag& b, v8f c) {
    // D = A(16x32 f16) * B(32x16 f16) + C(16x16 f32)
    return __builtin_amdgcn_wmma_f32_16x16x32_f16(
        /*neg_a=*/false, a.v, /*neg_b=*/false, b.v,
        /*c_mod=*/(short)0, c, /*reuse_a=*/false, /*reuse_b=*/false);
}

// ---------------------------------------------------------------------------
// CDNA5 async global->LDS copies (ASYNCcnt-tracked, no VGPR round trip).
// Per ISA the instruction offset is added to both the LDS address
//          and the global address (SADDR=off),
// so one address pair serves both 16B halves of a 32B copy.
// ---------------------------------------------------------------------------
static __device__ __forceinline__ void async_copy32(unsigned lds_off, const void* gaddr) {
    asm volatile("global_load_async_to_lds_b128 %0, %1, off\n\t"
                 "global_load_async_to_lds_b128 %0, %1, off offset:16"
                 :
                 : "v"(lds_off), "v"((unsigned long long)(uintptr_t)gaddr)
                 : "memory");
}
static __device__ __forceinline__ void wait_async0() {
    asm volatile("s_wait_asynccnt 0" ::: "memory");
}
// Low 32 bits of a generic LDS pointer are the wave-relative LDS byte offset.
static __device__ __forceinline__ unsigned lds_off(const void* p) {
    return (unsigned)(uintptr_t)p;
}

// ---------------------------------------------------------------------------
// Fold LoRA into effective weight:  Weff[n,k] = W[n,k] + (1/R) * sum_r A[k,r]*B[r,n]
// W: [N,K] f32 row-major, A: [K,R] f32, B: [R,N] f32 -> Weff: [N,K] f16
// ---------------------------------------------------------------------------
__global__ __launch_bounds__(256) void prep_weight(
    const float* __restrict__ W, const float* __restrict__ A,
    const float* __restrict__ Bm, _Float16* __restrict__ Weff, int N, int K)
{
    size_t i = (size_t)blockIdx.x * 256 + threadIdx.x;
    if (i >= (size_t)N * K) return;
    int n = (int)(i / K);
    int k = (int)(i % K);
    float acc = 0.f;
#pragma unroll
    for (int r = 0; r < RR; ++r)
        acc += A[(size_t)k * RR + r] * Bm[(size_t)r * N + n];
    Weff[i] = (_Float16)(W[i] + acc * (1.0f / RR));
}

// ---------------------------------------------------------------------------
// f32 -> f16 convert, 4 elems/thread
// ---------------------------------------------------------------------------
__global__ __launch_bounds__(256) void cvt_f16(
    const float* __restrict__ x, _Float16* __restrict__ y, size_t n)
{
    size_t i = ((size_t)blockIdx.x * 256 + threadIdx.x) * 4;
    if (i >= n) return;
    float4 f = *(const float4*)(x + i);
    y[i + 0] = (_Float16)f.x;
    y[i + 1] = (_Float16)f.y;
    y[i + 2] = (_Float16)f.z;
    y[i + 3] = (_Float16)f.w;
}

// ---------------------------------------------------------------------------
// GEMM: C[M,N] = A[M,K] * Bw[N,K]^T  (A,Bw f16 row-major, f32 accumulate)
// Block tile 128x128, 8 waves (4x2), wave tile 32x64 (2x4 WMMA), K-step 32.
// Double-buffered LDS, async global->LDS staging overlapped with WMMA.
// Padded LDS stride 40 halfs (80B -> 20-bank row shift).
// ---------------------------------------------------------------------------
template <bool OUT_F32>
__global__ __launch_bounds__(256) void gemm_wmma(
    const _Float16* __restrict__ A, const _Float16* __restrict__ Bw,
    float* __restrict__ Cf, _Float16* __restrict__ Ch, int M, int N, int K)
{
    __shared__ __align__(16) _Float16 aT[2][128 * 40];
    __shared__ __align__(16) _Float16 bT[2][128 * 40];

    const int tid  = threadIdx.x;
    const int lane = tid & 31;
    const int wave = tid >> 5;
    const int m0   = blockIdx.x * 128;
    const int n0   = blockIdx.y * 128;
    const int wm   = wave & 3;      // 0..3 -> 32-row sub-tile
    const int wn   = wave >> 2;     // 0..1 -> 64-col sub-tile
    const int hl   = lane >> 4;     // half-wave id
    const int l16  = lane & 15;

    v8f acc[2][4] = {};

    // Staging map: each thread copies 32B of A and 32B of B per K-step
    const int srow = tid & 127;
    const int skc  = (tid >> 7) * 16;     // 0 or 16

    const _Float16* gA = A + (size_t)(m0 + srow) * K + skc;
    const _Float16* gB = Bw + (size_t)(n0 + srow) * K + skc;
    const unsigned laA0 = lds_off(&aT[0][srow * 40 + skc]);
    const unsigned laA1 = lds_off(&aT[1][srow * 40 + skc]);
    const unsigned laB0 = lds_off(&bT[0][srow * 40 + skc]);
    const unsigned laB1 = lds_off(&bT[1][srow * 40 + skc]);

    auto stage = [&](int kk, int buf) {
        async_copy32(buf ? laA1 : laA0, gA + kk);
        async_copy32(buf ? laB1 : laB0, gB + kk);
    };

    stage(0, 0);
    wait_async0();
    __syncthreads();

    const int nsteps = K / 32;
    for (int it = 0; it < nsteps; ++it) {
        const int cur = it & 1;
        if (it + 1 < nsteps)
            stage((it + 1) * 32, cur ^ 1);   // DMA next tile while computing

        const _Float16* aC = aT[cur];
        const _Float16* bC = bT[cur];

        // A fragments: lane holds M = l16, K in [kb,kb+8) and [kb+16,kb+24)
        Frag fa[2], fb[4];
        const int akb = hl * 8;
#pragma unroll
        for (int mi = 0; mi < 2; ++mi) {
            const _Float16* p = aC + (wm * 32 + mi * 16 + l16) * 40 + akb;
            fa[mi].u[0] = *(const uint4*)(p);
            fa[mi].u[1] = *(const uint4*)(p + 16);
        }
        // B fragments: lane holds N = l16, K in [kb,kb+8) and [kb+16,kb+24)
        const int bkb = hl * 8;
#pragma unroll
        for (int ni = 0; ni < 4; ++ni) {
            const _Float16* p = bC + (wn * 64 + ni * 16 + l16) * 40 + bkb;
            fb[ni].u[0] = *(const uint4*)(p);
            fb[ni].u[1] = *(const uint4*)(p + 16);
        }
#pragma unroll
        for (int mi = 0; mi < 2; ++mi)
#pragma unroll
            for (int ni = 0; ni < 4; ++ni)
                acc[mi][ni] = wmma16(fa[mi], fb[ni], acc[mi][ni]);

        wait_async0();        // next tile resident
        __syncthreads();      // everyone done reading current buffer
    }

    // Epilogue: C layout lane -> N = l16, M = v + 8*hl
#pragma unroll
    for (int mi = 0; mi < 2; ++mi) {
#pragma unroll
        for (int ni = 0; ni < 4; ++ni) {
            const int col = n0 + wn * 64 + ni * 16 + l16;
#pragma unroll
            for (int v = 0; v < 8; ++v) {
                const int row = m0 + wm * 32 + mi * 16 + v + hl * 8;
                if constexpr (OUT_F32)
                    Cf[(size_t)row * N + col] = acc[mi][ni][v];
                else
                    Ch[(size_t)row * N + col] = (_Float16)acc[mi][ni][v];
            }
        }
    }
}

// ---------------------------------------------------------------------------
// Flash attention: one block = (128 q rows, one head, one batch); 8 waves,
// each wave owns 16 q rows. Key tiles of 32, online softmax, double-buffered
// K/V staging: K via async DMA, V transposed through VGPRs with its global
// loads issued before the current tile's compute and its LDS stores after.
// qkv layout: [B,S,3,H,HD] f16 (row stride N3). Output: [B,S,D] f16.
// ---------------------------------------------------------------------------
__global__ __launch_bounds__(256) void flash_attn(
    const _Float16* __restrict__ qkv, const float* __restrict__ alibi,
    const float* __restrict__ pad, _Float16* __restrict__ out)
{
    __shared__ __align__(16) _Float16 kT[2][32 * 136];   // [key][hd], padded
    __shared__ __align__(16) _Float16 vT[2][128 * 40];   // [hd][key], padded
    __shared__ __align__(16) _Float16 pbuf[8 * 16 * 40]; // per-wave P staging

    const int tid  = threadIdx.x;
    const int lane = tid & 31;
    const int wave = tid >> 5;
    const int b    = blockIdx.z;
    const int h    = blockIdx.y;
    const int q0   = blockIdx.x * 128 + wave * 16;   // this wave's first q row
    const int hl   = lane >> 4;
    const int l16  = lane & 15;
    const float sc = 0.08838834764831845f;           // 1/sqrt(128)

    // Load Q fragments once (4 chunks of K=32 over HD=128)
    Frag fq[4];
    {
        const int row = q0 + l16;
        const size_t base = ((size_t)(b * SS + row)) * N3 + (size_t)h * HDD;
        const int kb = hl * 8;
#pragma unroll
        for (int c = 0; c < 4; ++c) {
            fq[c].u[0] = *(const uint4*)(qkv + base + c * 32 + kb);
            fq[c].u[1] = *(const uint4*)(qkv + base + c * 32 + kb + 16);
        }
    }

    v8f accO[8] = {};
    float rm[8], rl[8];
#pragma unroll
    for (int v = 0; v < 8; ++v) { rm[v] = -INFINITY; rl[v] = 0.f; }

    _Float16* pmine = pbuf + wave * 16 * 40;

    // Staging map: 256 threads cover 32 keys x 128 hd (16 halfs each)
    const int skey = tid >> 3;        // key 0..31
    const int shd0 = (tid & 7) * 16;  // hd 0..112
    const size_t krow0 = ((size_t)(b * SS + skey)) * N3 + DD + (size_t)h * HDD + shd0;
    const size_t vrow0 = krow0 + DD;  // V plane right after K plane
    const unsigned lK0 = lds_off(&kT[0][skey * 136 + shd0]);
    const unsigned lK1 = lds_off(&kT[1][skey * 136 + shd0]);

    union VV { uint4 u[2]; _Float16 e[16]; };

    // Prologue: stage tile 0 (K async, V through VGPRs)
    {
        async_copy32(lK0, qkv + krow0);
        VV vv;
        vv.u[0] = *(const uint4*)(qkv + vrow0);
        vv.u[1] = *(const uint4*)(qkv + vrow0 + 8);
#pragma unroll
        for (int j = 0; j < 16; ++j)
            vT[0][(shd0 + j) * 40 + skey] = vv.e[j];
    }
    wait_async0();
    __syncthreads();

    const int ntiles = SS / 32;
    for (int it = 0; it < ntiles; ++it) {
        const int kt  = it * 32;
        const int cur = it & 1;
        const bool more = (it + 1 < ntiles);

        // Kick off next tile early: K DMA + V global loads (stores deferred)
        VV vv;
        if (more) {
            const size_t step = (size_t)(kt + 32) * N3;
            async_copy32(cur ? lK0 : lK1, qkv + krow0 + step);
            vv.u[0] = *(const uint4*)(qkv + vrow0 + step);
            vv.u[1] = *(const uint4*)(qkv + vrow0 + step + 8);
        }

        const _Float16* kC = kT[cur];
        const _Float16* vC = vT[cur];

        // S = Q K^T for 32 keys -> two 16x16 f32 tiles
        v8f s0 = {}, s1 = {};
        {
            const int kb2 = hl * 8;
#pragma unroll
            for (int c = 0; c < 4; ++c) {
                Frag f0, f1;
                const _Float16* p0 = kC + l16 * 136 + c * 32 + kb2;        // keys kt+0..15
                const _Float16* p1 = kC + (16 + l16) * 136 + c * 32 + kb2; // keys kt+16..31
                f0.u[0] = *(const uint4*)(p0); f0.u[1] = *(const uint4*)(p0 + 16);
                f1.u[0] = *(const uint4*)(p1); f1.u[1] = *(const uint4*)(p1 + 16);
                s0 = wmma16(fq[c], f0, s0);
                s1 = wmma16(fq[c], f1, s1);
            }
        }

        // Bias + online softmax (row reductions across 16-lane halves)
        const float pd0 = pad[(size_t)b * SS + kt + l16];
        const float pd1 = pad[(size_t)b * SS + kt + 16 + l16];
#pragma unroll
        for (int v = 0; v < 8; ++v) {
            const int row = q0 + v + hl * 8;
            const size_t abase = ((size_t)h * SS + row) * SS + kt;
            float x0 = s0[v] * sc + alibi[abase + l16] + pd0;
            float x1 = s1[v] * sc + alibi[abase + 16 + l16] + pd1;
            float mx = fmaxf(x0, x1);
#pragma unroll
            for (int off = 1; off < 16; off <<= 1)
                mx = fmaxf(mx, __shfl_xor(mx, off, 32));
            const float mnew  = fmaxf(rm[v], mx);
            const float scale = __expf(rm[v] - mnew);
            const float e0 = __expf(x0 - mnew);
            const float e1 = __expf(x1 - mnew);
            float rs = e0 + e1;
#pragma unroll
            for (int off = 1; off < 16; off <<= 1)
                rs += __shfl_xor(rs, off, 32);
            rl[v] = rl[v] * scale + rs;
            rm[v] = mnew;
#pragma unroll
            for (int ht = 0; ht < 8; ++ht)
                accO[ht][v] *= scale;
            // Stage P (C layout -> row-major [16 x 32] in wave-private LDS)
            pmine[(v + hl * 8) * 40 + l16]      = (_Float16)e0;
            pmine[(v + hl * 8) * 40 + 16 + l16] = (_Float16)e1;
        }
        __builtin_amdgcn_wave_barrier();

        // Reload P as A-fragment (16x32), then O += P * V
        Frag fp;
        {
            const _Float16* p = pmine + l16 * 40 + hl * 8;
            fp.u[0] = *(const uint4*)(p);
            fp.u[1] = *(const uint4*)(p + 16);
        }
        const int kb2 = hl * 8;
#pragma unroll
        for (int ht = 0; ht < 8; ++ht) {
            Frag fv;
            const _Float16* p = vC + (ht * 16 + l16) * 40 + kb2;
            fv.u[0] = *(const uint4*)(p);
            fv.u[1] = *(const uint4*)(p + 16);
            accO[ht] = wmma16(fp, fv, accO[ht]);
        }

        // Deferred V-transpose stores for the next tile (other buffer)
        if (more) {
            _Float16* vN = vT[cur ^ 1];
#pragma unroll
            for (int j = 0; j < 16; ++j)
                vN[(shd0 + j) * 40 + skey] = vv.e[j];
        }

        wait_async0();        // next K tile resident
        __syncthreads();      // all waves done with current buffers
    }

    // Normalize and store f16 output [B,S,D]
#pragma unroll
    for (int ht = 0; ht < 8; ++ht) {
#pragma unroll
        for (int v = 0; v < 8; ++v) {
            const int row = q0 + v + hl * 8;
            const float o = accO[ht][v] / rl[v];
            out[((size_t)(b * SS + row)) * DD + (size_t)h * HDD + ht * 16 + l16] = (_Float16)o;
        }
    }
}

// ---------------------------------------------------------------------------
// Host-side launcher
// ---------------------------------------------------------------------------
extern "C" void kernel_launch(void* const* d_in, const int* in_sizes, int n_in,
                              void* d_out, int out_size, void* d_ws, size_t ws_size,
                              hipStream_t stream)
{
    (void)in_sizes; (void)n_in; (void)out_size; (void)ws_size;

    const float* x     = (const float*)d_in[0];
    const float* alibi = (const float*)d_in[1];
    const float* pad   = (const float*)d_in[2];
    const float* Wqkv  = (const float*)d_in[3];
    const float* Aqkv  = (const float*)d_in[4];
    const float* Bqkv  = (const float*)d_in[5];
    const float* Wout  = (const float*)d_in[6];
    const float* Aout  = (const float*)d_in[7];
    const float* Bout  = (const float*)d_in[8];
    float* out = (float*)d_out;

    // Workspace layout (f16 buffers), 256B aligned sections
    char* ws = (char*)d_ws;
    auto take = [&](size_t bytes) {
        char* p = ws;
        ws += (bytes + 255) & ~(size_t)255;
        return p;
    };
    _Float16* x16   = (_Float16*)take((size_t)MM * DD * 2);   // [M,D]
    _Float16* wq16  = (_Float16*)take((size_t)N3 * DD * 2);   // [3D,D] effective
    _Float16* wo16  = (_Float16*)take((size_t)DD * DD * 2);   // [D,D] effective
    _Float16* qkv16 = (_Float16*)take((size_t)MM * N3 * 2);   // [M,3D]
    _Float16* ao16  = (_Float16*)take((size_t)MM * DD * 2);   // attention out [M,D]

    // 1) Fold LoRA into effective weights
    {
        size_t nq = (size_t)N3 * DD;
        prep_weight<<<(unsigned)((nq + 255) / 256), 256, 0, stream>>>(
            Wqkv, Aqkv, Bqkv, wq16, N3, DD);
        size_t no = (size_t)DD * DD;
        prep_weight<<<(unsigned)((no + 255) / 256), 256, 0, stream>>>(
            Wout, Aout, Bout, wo16, DD, DD);
    }
    // 2) Convert activations
    {
        size_t nx = (size_t)MM * DD;
        cvt_f16<<<(unsigned)(nx / 4 / 256), 256, 0, stream>>>(x, x16, nx);
    }
    // 3) QKV projection: qkv16[M,3D] = x16 * wq16^T
    {
        dim3 g(MM / 128, N3 / 128);
        gemm_wmma<false><<<g, 256, 0, stream>>>(x16, wq16, nullptr, qkv16, MM, N3, DD);
    }
    // 4) Fused flash attention with ALiBi bias + padding mask
    {
        dim3 g(SS / 128, HH, BN);
        flash_attn<<<g, 256, 0, stream>>>(qkv16, alibi, pad, ao16);
    }
    // 5) Output projection -> f32 d_out
    {
        dim3 g(MM / 128, DD / 128);
        gemm_wmma<true><<<g, 256, 0, stream>>>(ao16, wo16, out, nullptr, MM, DD, DD);
    }
}
